// VQVAEv4_50337016709444
// MI455X (gfx1250) — matrix operand order repaired and checked by reading, not compiled, not run on hardware
//
#include <hip/hip_runtime.h>
#include <hip/hip_bf16.h>

// ---------------------------------------------------------------------------
// VQ-VAE 3D forward for gfx1250 (MI455X). All GEMM-like work (all convs,
// transposed convs, VQ distance matrix) runs through v_wmma_f32_16x16x32_f16
// with f32 accumulation. Wave32; block tile = 16 spatial x 128 channels
// (8 waves x one 16x16 WMMA tile each); A fragment staged in LDS and shared
// by all 8 waves; weights read as consecutive float4 (w is [Cout][Ktot]).
// ---------------------------------------------------------------------------

typedef __attribute__((ext_vector_type(16))) _Float16 v16h;
typedef __attribute__((ext_vector_type(8)))  float    v8f;

#define EPS 1e-5f

// ---------------- input index map (setup_inputs dict order, flattened) -----
enum {
  IN_X = 0,
  IN_E1W, IN_E1B, IN_E1BN_G, IN_E1BN_B, IN_E1BN_M, IN_E1BN_V,       // 1..6
  IN_E2W, IN_E2B, IN_E2BN_G, IN_E2BN_B, IN_E2BN_M, IN_E2BN_V,       // 7..12
  IN_ER1 = 13,                 // res block: g1,b1,m1,v1,w1,bb1,g2,b2,m2,v2,w2,bb2
  IN_ER2 = IN_ER1 + 12,        // 25
  IN_ETW = IN_ER2 + 12,        // 37
  IN_ETB,                      // 38
  IN_CODEBOOK,                 // 39
  IN_DR1,                      // 40
  IN_DR2 = IN_DR1 + 12,        // 52
  IN_DR3 = IN_DR2 + 12,        // 64
  IN_U1W = IN_DR3 + 12,        // 76
  IN_U1B, IN_U1BN_G, IN_U1BN_B, IN_U1BN_M, IN_U1BN_V,               // 77..81
  IN_DR4,                      // 82
  IN_U2W = IN_DR4 + 12,        // 94
  IN_U2B, IN_U2BN_G, IN_U2BN_B, IN_U2BN_M, IN_U2BN_V,               // 95..99
  IN_OW,                       // 100
  IN_OB                        // 101
};

// ---------------- small helper kernels -------------------------------------
__global__ void k_zero(float* p, int n) {
  int i = blockIdx.x * blockDim.x + threadIdx.x;
  if (i < n) p[i] = 0.0f;
}

// fold BN into per-channel scale/shift applied after conv bias:
// y = (conv + bias) * scale + shift ; scale = g/sqrt(v+eps) ; shift = b - m*scale
__global__ void k_scale_shift(const float* __restrict__ g, const float* __restrict__ b,
                              const float* __restrict__ m, const float* __restrict__ v,
                              float* __restrict__ scale, float* __restrict__ shift, int C) {
  int i = blockIdx.x * blockDim.x + threadIdx.x;
  if (i < C) {
    float s = g[i] * rsqrtf(v[i] + EPS);
    scale[i] = s;
    shift[i] = b[i] - m[i] * s;
  }
}

// pre-activation: y = relu((x - m) * g/sqrt(v+eps) + b), NC(DHW) layout
__global__ void k_bn_relu(const float* __restrict__ x,
                          const float* __restrict__ g, const float* __restrict__ b,
                          const float* __restrict__ m, const float* __restrict__ v,
                          float* __restrict__ y, int C, int S) {
  int i = blockIdx.x * blockDim.x + threadIdx.x;
  if (i < C * S) {
    int c = i / S;
    float s = g[c] * rsqrtf(v[c] + EPS);
    float t = (x[i] - m[c]) * s + b[c];
    y[i] = fmaxf(t, 0.0f);
  }
}

// ---------------- generic implicit-GEMM conv3d via WMMA --------------------
// mode 0: forward conv, cubic Din^3 -> Dout^3, stride/pad as usual.
// mode 1: ConvTranspose3d(k,s=2,p=1) via lhs-dilation identity: pass
//         pad = K-1-p (=2); tap valid iff q = o + k - pad is even and q/2 in range.
// Requires: Mtot % 16 == 0, Ktot % 32 == 0 (true for every layer here).
// Block: 256 thr = 8 waves. Tile: M=16 (shared A, staged in LDS) x N=128.
__global__ __launch_bounds__(256)
void k_conv_wmma(const float* __restrict__ x, const float* __restrict__ w,
                 const float* __restrict__ bias, const float* __restrict__ scale,
                 const float* __restrict__ shift, const float* __restrict__ resid,
                 float* __restrict__ y,
                 int Cin, int Cout, int Din, int Dout,
                 int Kk, int stride, int pad, int mode, int relu) {
  __shared__ __align__(32) _Float16 ldsA[512];  // 16x32 A tile, fragment layout
  __shared__ int4 tap[32];                      // per-chunk (ciBase, kd, kh, kw)

  const int t    = threadIdx.x;
  const int lane = t & 31;
  const int wave = t >> 5;
  const int Mtot = Dout * Dout * Dout;
  const int m0   = blockIdx.x * 16;
  const int n0   = blockIdx.y * 128 + wave * 16;
  const int hf   = lane >> 4;                   // wave half (A/B K split)
  const int lo   = lane & 15;
  const int KK = Kk * Kk, KKK = KK * Kk;
  const int Ktot = Cin * KKK;
  const int DD = Din * Din;

  // staging role: thread t fills fragment position e of A rows (t>>4) for
  // both wave-halves -> LDS slots t and t+256.
  const int ms = m0 + (t >> 4);
  int soz, soy, sox;
  { int q = ms; soz = q / (Dout * Dout); q -= soz * Dout * Dout;
    soy = q / Dout; sox = q - soy * Dout; }
  const int e  = t & 15;
  const int p0 = e;                             // K-perm for half 0
  const int p1 = p0 + 16;                       // K-perm for half 1

  const int nc   = n0 + lo;                     // this lane's output channel
  const bool nok = nc < Cout;

  v8f c = {};
  for (int k0 = 0; k0 < Ktot; k0 += 32) {
    // -- per-chunk tap table: hoist the 3 integer divisions out of the gather
    if (t < 32) {
      int k = k0 + t;
      int ci = k / KKK; int r = k - ci * KKK;
      int kd = r / KK;  r -= kd * KK;
      int kh = r / Kk;  int kw = r - kh * Kk;
      tap[t] = make_int4(ci * DD * Din, kd, kh, kw);
    }
    __syncthreads();
    // -- stage A fragment (shared by all 8 waves)
#pragma unroll
    for (int s = 0; s < 2; ++s) {
      int4 tp = tap[s == 0 ? p0 : p1];
      float av = 0.0f;
      int iz, iy, ix; bool ok;
      if (mode == 0) {
        iz = soz * stride - pad + tp.y;
        iy = soy * stride - pad + tp.z;
        ix = sox * stride - pad + tp.w;
        ok = ((unsigned)iz < (unsigned)Din) & ((unsigned)iy < (unsigned)Din) &
             ((unsigned)ix < (unsigned)Din);
      } else {
        int qz = soz + tp.y - pad, qy = soy + tp.z - pad, qx = sox + tp.w - pad;
        ok = ((qz | qy | qx) >= 0) && (((qz | qy | qx) & (stride - 1)) == 0);
        iz = qz / stride; iy = qy / stride; ix = qx / stride;
        ok = ok && (iz < Din) && (iy < Din) && (ix < Din);
      }
      if (ok) av = x[(size_t)tp.x + iz * DD + iy * Din + ix];
      ldsA[t + s * 256] = (_Float16)av;
    }
    __syncthreads();
    // -- consume: A from LDS (2x ds_load_b128), B as 4x float4 from global
    v16h a = *(const v16h*)&ldsA[lane * 16];
    v16h b = {};
    if (nok) {
      const float4* wp = (const float4*)(w + (size_t)nc * Ktot + k0 + hf * 16);
      if (k0 + 32 < Ktot) __builtin_prefetch(wp + 8, 0, 0);  // global_prefetch_b8
#pragma unroll
      for (int q = 0; q < 4; ++q) {
        float4 f = wp[q];
        b[q * 4 + 0] = (_Float16)f.x; b[q * 4 + 1] = (_Float16)f.y;
        b[q * 4 + 2] = (_Float16)f.z; b[q * 4 + 3] = (_Float16)f.w;
      }
    }
    c = __builtin_amdgcn_wmma_f32_16x16x32_f16(false, a, false, b, (short)0, c,
                                               false, false);
  }
  // D layout: vgpr r -> M = m0 + r + 8*hf ; N = n0 + lo
#pragma unroll
  for (int r = 0; r < 8; ++r) {
    int mo = m0 + r + 8 * hf;
    if (nok) {
      float v = c[r];
      if (bias)  v += bias[nc];
      if (scale) v *= scale[nc];
      if (shift) v += shift[nc];
      if (relu)  v = fmaxf(v, 0.0f);
      if (resid) v += resid[(size_t)nc * Mtot + mo];
      y[(size_t)nc * Mtot + mo] = v;
    }
  }
}

// ---------------- VQ: S = z @ E^T via WMMA (M=N=512, K=256) ----------------
// z[m][c] = z_e[c*512+m] (channel-major activation), E[n][c] row-major codebook.
__global__ __launch_bounds__(256)
void k_vq_gemm(const float* __restrict__ ze, const float* __restrict__ E,
               float* __restrict__ Sout) {
  const int lane = threadIdx.x & 31;
  const int wave = threadIdx.x >> 5;
  const int m0 = (blockIdx.x * 8 + wave) * 16;
  const int n0 = blockIdx.y * 16;
  const int hf = lane >> 4, lo = lane & 15;
  v8f c = {};
  for (int k0 = 0; k0 < 256; k0 += 32) {
    v16h a, b;
#pragma unroll
    for (int e = 0; e < 16; ++e) {
      int kA = k0 + hf * 16 + e;
      a[e] = (_Float16)ze[kA * 512 + (m0 + lo)];
    }
    const float4* ep = (const float4*)(E + (n0 + lo) * 256 + k0 + hf * 16);
#pragma unroll
    for (int q = 0; q < 4; ++q) {
      float4 f = ep[q];
      b[q * 4 + 0] = (_Float16)f.x; b[q * 4 + 1] = (_Float16)f.y;
      b[q * 4 + 2] = (_Float16)f.z; b[q * 4 + 3] = (_Float16)f.w;
    }
    c = __builtin_amdgcn_wmma_f32_16x16x32_f16(false, a, false, b, (short)0, c,
                                               false, false);
  }
#pragma unroll
  for (int r = 0; r < 8; ++r)
    Sout[(m0 + r + 8 * hf) * 512 + (n0 + lo)] = c[r];
}

__global__ void k_enorm(const float* __restrict__ E, float* __restrict__ en) {
  int n = blockIdx.x * blockDim.x + threadIdx.x;
  if (n < 512) {
    float s = 0.0f;
    for (int c = 0; c < 256; ++c) { float e = E[n * 256 + c]; s += e * e; }
    en[n] = s;
  }
}

// one block per latent point: argmin over 512 codes (||z||^2 term is a row
// constant -> dropped), gather z_q, accumulate commitment sum via atomics.
__global__ __launch_bounds__(256)
void k_vq_assign(const float* __restrict__ S, const float* __restrict__ en,
                 const float* __restrict__ E, const float* __restrict__ ze,
                 float* __restrict__ zq, float* __restrict__ idx_out,
                 float* __restrict__ acc) {
  __shared__ float sd[256];
  __shared__ int   si[256];
  const int m = blockIdx.x;
  const int t = threadIdx.x;
  float best = 3.4e38f; int bi = 0x7fffffff;
  for (int n = t; n < 512; n += 256) {
    float d = en[n] - 2.0f * S[m * 512 + n];
    if (d < best) { best = d; bi = n; }
  }
  sd[t] = best; si[t] = bi;
  __syncthreads();
  for (int s = 128; s > 0; s >>= 1) {
    if (t < s) {
      if (sd[t + s] < sd[t] || (sd[t + s] == sd[t] && si[t + s] < si[t])) {
        sd[t] = sd[t + s]; si[t] = si[t + s];
      }
    }
    __syncthreads();
  }
  const int code = si[0];
  if (t == 0) idx_out[m] = (float)code;   // jnp concat upcasts idx to f32
  float part = 0.0f;
  for (int c = t; c < 256; c += 256) {
    float q = E[code * 256 + c];
    zq[c * 512 + m] = q;                  // forward value of straight-through z_q
    float d = q - ze[c * 512 + m];
    part += d * d;
  }
  __syncthreads();
  sd[t] = part;
  __syncthreads();
  for (int s = 128; s > 0; s >>= 1) {
    if (t < s) sd[t] += sd[t + s];
    __syncthreads();
  }
  if (t == 0) atomicAdd(acc, sd[0]);
}

__global__ void k_finalize(const float* __restrict__ acc, float* __restrict__ out) {
  if (threadIdx.x == 0 && blockIdx.x == 0)
    out[0] = 0.25f * acc[0] / 131072.0f;  // COMMITMENT_COST * mean
}

// ---------------------------------------------------------------------------
extern "C" void kernel_launch(void* const* d_in, const int* in_sizes, int n_in,
                              void* d_out, int out_size, void* d_ws, size_t ws_size,
                              hipStream_t stream) {
  (void)n_in; (void)out_size; (void)ws_size;
  auto F = [&](int i) { return (const float*)d_in[i]; };

  // workspace layout (floats): ~20.5 MB total
  float* ws   = (float*)d_ws;
  float* buf0 = ws;                    // 3,145,728 (max: 96 @ 32^3)
  float* buf1 = buf0 + 3145728;        //   786,432 (max: 192 @ 16^3)
  float* buf2 = buf1 + 786432;         //   786,432
  float* ze   = buf2 + 786432;         //   131,072 (z_e: 256 @ 8^3)
  float* Smat = ze   + 131072;         //   262,144 (z @ E^T)
  float* en   = Smat + 262144;         //       512
  float* psc  = en   + 512;            //      4096 (per-channel scale)
  float* psh  = psc  + 4096;           //      4096 (per-channel shift)
  float* acc  = psh  + 4096;           //        16 (commit accumulator)

  float* out        = (float*)d_out;
  const size_t OUT_COMMIT = (size_t)3717 * 32768;  // 121,798,656
  float* out_commit = out + OUT_COMMIT;
  float* out_idx    = out_commit + 1;              // 512 floats

  auto mkss = [&](int base) {          // base -> g,b,m,v at d_in[base..base+3]
    int C = in_sizes[base];
    k_scale_shift<<<(C + 255) / 256, 256, 0, stream>>>(F(base), F(base + 1),
                                                       F(base + 2), F(base + 3),
                                                       psc, psh, C);
  };
  auto conv = [&](const float* in, const float* wgt, const float* bias,
                  const float* sc, const float* sh, const float* resid, float* outp,
                  int Cin, int Cout, int Din, int Dout, int K, int stride, int pad,
                  int mode, int relu) {
    int M = Dout * Dout * Dout;                    // always a multiple of 16
    dim3 grid(M / 16, (Cout + 127) / 128);
    k_conv_wmma<<<grid, 256, 0, stream>>>(in, wgt, bias, sc, sh, resid, outp,
                                          Cin, Cout, Din, Dout, K, stride, pad,
                                          mode, relu);
  };
  auto bnrelu = [&](const float* xin, int base, float* outp, int C, int Ssp) {
    int n = C * Ssp;
    k_bn_relu<<<(n + 255) / 256, 256, 0, stream>>>(xin, F(base), F(base + 1),
                                                   F(base + 2), F(base + 3),
                                                   outp, C, Ssp);
  };
  // pre-activation residual block, in-place on xbuf (per-thread read-then-write)
  auto resblock = [&](int base, float* xbuf, float* tA, float* tB, int C, int D) {
    int Ssp = D * D * D;
    bnrelu(xbuf, base + 0, tA, C, Ssp);
    conv(tA, F(base + 4), F(base + 5), nullptr, nullptr, nullptr, tB,
         C, C, D, D, 3, 1, 1, 0, 0);
    bnrelu(tB, base + 6, tA, C, Ssp);
    conv(tA, F(base + 10), F(base + 11), nullptr, nullptr, xbuf, xbuf,
         C, C, D, D, 3, 1, 1, 0, 0);
  };

  // ---------------- encoder ----------------
  mkss(IN_E1BN_G);
  conv(F(IN_X), F(IN_E1W), F(IN_E1B), psc, psh, nullptr, buf0,
       32, 96, 32, 16, 4, 2, 1, 0, 1);                       // 96 @ 16^3
  mkss(IN_E2BN_G);
  conv(buf0, F(IN_E2W), F(IN_E2B), psc, psh, nullptr, buf1,
       96, 192, 16, 8, 4, 2, 1, 0, 1);                       // 192 @ 8^3
  resblock(IN_ER1, buf1, buf2, buf0, 192, 8);
  resblock(IN_ER2, buf1, buf2, buf0, 192, 8);
  conv(buf1, F(IN_ETW), F(IN_ETB), nullptr, nullptr, nullptr, ze,
       192, 256, 8, 8, 3, 1, 1, 0, 0);                       // z_e: 256 @ 8^3

  // ---------------- vector quantization ----------------
  k_zero<<<1, 32, 0, stream>>>(acc, 16);
  k_enorm<<<2, 256, 0, stream>>>(F(IN_CODEBOOK), en);
  { dim3 g(4, 32); k_vq_gemm<<<g, 256, 0, stream>>>(ze, F(IN_CODEBOOK), Smat); }
  k_vq_assign<<<512, 256, 0, stream>>>(Smat, en, F(IN_CODEBOOK), ze, buf0,
                                       out_idx, acc);        // z_q -> buf0

  // ---------------- decoder ----------------
  resblock(IN_DR1, buf0, buf1, buf2, 256, 8);
  resblock(IN_DR2, buf0, buf1, buf2, 256, 8);
  resblock(IN_DR3, buf0, buf1, buf2, 256, 8);
  mkss(IN_U1BN_G);
  conv(buf0, F(IN_U1W), F(IN_U1B), psc, psh, nullptr, buf1,  // convT k4 s2 p1
       256, 192, 8, 16, 4, 2, /*padT=*/2, 1, 1);             // 192 @ 16^3
  resblock(IN_DR4, buf1, buf2, buf0, 192, 16);
  mkss(IN_U2BN_G);
  conv(buf1, F(IN_U2W), F(IN_U2B), psc, psh, nullptr, buf0,  // convT k4 s2 p1
       192, 96, 16, 32, 4, 2, /*padT=*/2, 1, 1);             // 96 @ 32^3
  conv(buf0, F(IN_OW), F(IN_OB), nullptr, nullptr, nullptr, out,
       96, 3717, 32, 32, 3, 1, 1, 0, 0);                     // logits -> d_out

  k_finalize<<<1, 1, 0, stream>>>(acc, out_commit);
}
